// SRNN_ALIF_v2_87050397155443
// MI455X (gfx1250) — matrix operand order repaired and checked by reading, not compiled, not run on hardware
//
#include <hip/hip_runtime.h>

typedef __bf16 bf16;
typedef __attribute__((ext_vector_type(16))) __bf16 v16bf;
typedef __attribute__((ext_vector_type(8)))  __bf16 v8bf;
typedef __attribute__((ext_vector_type(8)))  float  v8f;

#define HDIM 512
#define T_STEPS 98

// ---------------- weight conversion: fp32 -> bf16 (RNE via compiler cast) ----------------
__global__ void cvt_bf16_kernel(const float* __restrict__ src, bf16* __restrict__ dst, int n) {
    int i = blockIdx.x * blockDim.x + threadIdx.x;
    if (i < n) dst[i] = (bf16)src[i];
}

// A fragment (16x32 bf16, MxK) from LDS spike buffer, row-major [16][512].
// Contiguous K order (same as B): lane (m = lane&15, g = lane>>4) holds K run [k0+16g, +16).
__device__ __forceinline__ v16bf lda_frag(const bf16* S, int m, int g, int k0) {
    const bf16* base = S + m * HDIM + k0 + g * 16;
    v8bf lo = *(const v8bf*)(base);
    v8bf hi = *(const v8bf*)(base + 8);
    return __builtin_shufflevector(lo, hi, 0,1,2,3,4,5,6,7,8,9,10,11,12,13,14,15);
}

__device__ __forceinline__ v8f wmma_bf16(v16bf a, v16bf b, v8f c) {
    return __builtin_amdgcn_wmma_f32_16x16x32_bf16(false, a, false, b, (short)0, c, false, false);
}

// Fused two-tile GEMM over one A source.  W is a wave-uniform base (stays in SGPRs);
// offA/offB are the per-lane 32-bit row offsets (global_load saddr+voffset form, k0
// folds into the 24-bit immediate).  One A fragment feeds two WMMAs; B fragments are
// double-buffered one k-step ahead to overlap L2 latency with the matrix pipe.
__device__ __forceinline__ void gemm_pair(v8f& c0, v8f& c1, const bf16* S,
                                          const bf16* __restrict__ W,
                                          int offA, int offB, int m, int g) {
    v16bf b0 = *(const v16bf*)(W + offA);
    v16bf b1 = *(const v16bf*)(W + offB);
#pragma unroll 4
    for (int k0 = 0; k0 < HDIM - 32; k0 += 32) {
        v16bf a   = lda_frag(S, m, g, k0);
        v16bf b0n = *(const v16bf*)(W + offA + k0 + 32);
        v16bf b1n = *(const v16bf*)(W + offB + k0 + 32);
        c0 = wmma_bf16(a, b0, c0);
        c1 = wmma_bf16(a, b1, c1);
        b0 = b0n; b1 = b1n;
    }
    v16bf a = lda_frag(S, m, g, HDIM - 32);
    c0 = wmma_bf16(a, b0, c0);
    c1 = wmma_bf16(a, b1, c1);
}

// Warm the WGP-level cache with the weight rows this lane will stream next
// (locality 3 -> near scope).
__device__ __forceinline__ void prefetch_rows(const bf16* __restrict__ W, int offA, int offB) {
    __builtin_prefetch(W + offA, 0, 3);
    __builtin_prefetch(W + offA + 256, 0, 3);
    __builtin_prefetch(W + offB, 0, 3);
    __builtin_prefetch(W + offB + 256, 0, 3);
}

// ALIF pointwise update for one 16x16 tile's 8 per-lane elements (column n, rows j+8g).
__device__ __forceinline__ void alif_tile(const float* inp, float al, float ro,
                                          float* mem, float* bb, bf16* S, int n, int g,
                                          float* sum /* nullable */) {
#pragma unroll
    for (int j = 0; j < 8; ++j) {
        int mm = j + 8 * g;
        float so  = (float)S[mm * HDIM + n];              // previous spike (0/1, exact)
        float b   = ro * bb[j] + (1.f - ro) * so;
        float Bth = 0.01f + 1.8f * b;
        float mv  = mem[j] * al + (1.f - al) * inp[j] - Bth * so;
        float sp  = (mv - Bth) > 0.f ? 1.f : 0.f;
        mem[j] = mv;
        bb[j]  = b;
        S[mm * HDIM + n] = (bf16)sp;
        if (sum) sum[j] += sp;
    }
}

__global__ __launch_bounds__(512, 1) void srnn_alif_kernel(
    const float* __restrict__ x,
    const float* __restrict__ i2h1_w, const float* __restrict__ i2h1_b,
    const float* __restrict__ h2h1_b,
    const float* __restrict__ i2h2_b, const float* __restrict__ h2h2_b,
    const float* __restrict__ i2h3_b, const float* __restrict__ h2h3_b,
    const float* __restrict__ h2o_w,  const float* __restrict__ h2o_b,
    const float* __restrict__ tadp1, const float* __restrict__ tadp2, const float* __restrict__ tadp3,
    const float* __restrict__ tm1,   const float* __restrict__ tm2,   const float* __restrict__ tm3,
    const bf16* __restrict__ Wh1, const bf16* __restrict__ Wi2, const bf16* __restrict__ Wh2,
    const bf16* __restrict__ Wi3, const bf16* __restrict__ Wh3,
    float* __restrict__ out)
{
    __shared__ bf16 S1[16 * HDIM];   // spikes layer 1 (bf16, 16 KB)
    __shared__ bf16 S2[16 * HDIM];
    __shared__ bf16 S3[16 * HDIM];
    __shared__ float XW[16 * 8];     // per-step input window

    const int tid   = threadIdx.x;
    const int lane  = tid & 31;
    const int wave  = tid >> 5;      // 16 waves
    const int m     = lane & 15;     // batch row within tile (A-layout), column (B/C-layout)
    const int g     = lane >> 4;     // lane-half
    const int koffB = g * 16;        // B-fragment K offset for this lane-half
    const int nl    = wave * 32 + m; // hidden column of tile 0 (tile 1 = nl+16)
    const int row0  = blockIdx.x * 16;

    // single per-lane row offset pair shared by all five weight matrices
    const int offA = nl * HDIM + koffB;
    const int offB = (nl + 16) * HDIM + koffB;

    for (int i = tid; i < 16 * HDIM; i += 512) {
        S1[i] = (bf16)0.f; S2[i] = (bf16)0.f; S3[i] = (bf16)0.f;
    }

    // per-lane, per-tile constants and state (column = nl + 16*t, fixed per lane)
    float al1[2], ro1[2], bs1[2], al2[2], ro2[2], bs2[2], al3[2], ro3[2], bs3[2];
    v16bf bw1[2];   // zero-padded K=8 input-weight B fragments for layer 1
    float mem1[2][8], bb1[2][8], mem2[2][8], bb2[2][8], mem3[2][8], bb3[2][8], sum3[2][8];

#pragma unroll
    for (int t = 0; t < 2; ++t) {
        int n = nl + 16 * t;
        al1[t] = expf(-1.f / tm1[n]); ro1[t] = expf(-1.f / tadp1[n]); bs1[t] = i2h1_b[n] + h2h1_b[n];
        al2[t] = expf(-1.f / tm2[n]); ro2[t] = expf(-1.f / tadp2[n]); bs2[t] = i2h2_b[n] + h2h2_b[n];
        al3[t] = expf(-1.f / tm3[n]); ro3[t] = expf(-1.f / tadp3[n]); bs3[t] = i2h3_b[n] + h2h3_b[n];
        v16bf bw;
#pragma unroll
        for (int e = 0; e < 16; ++e) {
            int k = koffB + e;
            float v = (k < 8) ? i2h1_w[n * 8 + k] : 0.f;   // K padded 8 -> 32 with zeros
            bw[e] = (bf16)v;
        }
        bw1[t] = bw;
#pragma unroll
        for (int j = 0; j < 8; ++j) {
            mem1[t][j] = 0.f; bb1[t][j] = 0.01f;
            mem2[t][j] = 0.f; bb2[t][j] = 0.01f;
            mem3[t][j] = 0.f; bb3[t][j] = 0.01f;
            sum3[t][j] = 0.f;
        }
    }

    const v8f vzero = {0.f, 0.f, 0.f, 0.f, 0.f, 0.f, 0.f, 0.f};
    __syncthreads();

    for (int tt = 0; tt < T_STEPS; ++tt) {
        // replicate reference window slicing (clamped start)
        int start = (tt * 8 < 90) ? tt * 8 : 776;
        if (tid < 128) XW[tid] = x[(row0 + (tid >> 3)) * 784 + start + (tid & 7)];
        // (XW becomes visible at the barrier after the L1 recurrent GEMM)

        // ---------- Layer 1: recurrent GEMM on s1(t-1) ----------
        prefetch_rows(Wi2, offA, offB);
        v8f c10 = vzero, c11 = vzero;
        gemm_pair(c10, c11, S1, Wh1, offA, offB, m, g);
        __syncthreads();   // all S1 reads done; XW visible

        // input term: one zero-padded WMMA per tile (K=8 in K=32 slot)
        v16bf ax;
#pragma unroll
        for (int e = 0; e < 16; ++e) ax[e] = (bf16)0.f;
        if (g == 0) {
#pragma unroll
            for (int e = 0; e < 8; ++e) ax[e] = (bf16)XW[m * 8 + e];
        }
        c10 = wmma_bf16(ax, bw1[0], c10);
        c11 = wmma_bf16(ax, bw1[1], c11);

        {
            float inp[8];
#pragma unroll
            for (int j = 0; j < 8; ++j) inp[j] = c10[j] + bs1[0];
            alif_tile(inp, al1[0], ro1[0], mem1[0], bb1[0], S1, nl,      g, nullptr);
#pragma unroll
            for (int j = 0; j < 8; ++j) inp[j] = c11[j] + bs1[1];
            alif_tile(inp, al1[1], ro1[1], mem1[1], bb1[1], S1, nl + 16, g, nullptr);
        }
        __syncthreads();   // s1(t) ready

        // ---------- Layer 2: s1(t) @ Wi2^T + s2(t-1) @ Wh2^T ----------
        prefetch_rows(Wh2, offA, offB);
        v8f c20 = vzero, c21 = vzero;
        gemm_pair(c20, c21, S1, Wi2, offA, offB, m, g);
        prefetch_rows(Wi3, offA, offB);
        gemm_pair(c20, c21, S2, Wh2, offA, offB, m, g);
        __syncthreads();
        {
            float inp[8];
#pragma unroll
            for (int j = 0; j < 8; ++j) inp[j] = c20[j] + bs2[0];
            alif_tile(inp, al2[0], ro2[0], mem2[0], bb2[0], S2, nl,      g, nullptr);
#pragma unroll
            for (int j = 0; j < 8; ++j) inp[j] = c21[j] + bs2[1];
            alif_tile(inp, al2[1], ro2[1], mem2[1], bb2[1], S2, nl + 16, g, nullptr);
        }
        __syncthreads();   // s2(t) ready

        // ---------- Layer 3: s2(t) @ Wi3^T + s3(t-1) @ Wh3^T ----------
        prefetch_rows(Wh3, offA, offB);
        v8f c30 = vzero, c31 = vzero;
        gemm_pair(c30, c31, S2, Wi3, offA, offB, m, g);
        prefetch_rows(Wh1, offA, offB);
        gemm_pair(c30, c31, S3, Wh3, offA, offB, m, g);
        __syncthreads();
        {
            float inp[8];
#pragma unroll
            for (int j = 0; j < 8; ++j) inp[j] = c30[j] + bs3[0];
            alif_tile(inp, al3[0], ro3[0], mem3[0], bb3[0], S3, nl,      g, sum3[0]);
#pragma unroll
            for (int j = 0; j < 8; ++j) inp[j] = c31[j] + bs3[1];
            alif_tile(inp, al3[1], ro3[1], mem3[1], bb3[1], S3, nl + 16, g, sum3[1]);
        }
        __syncthreads();   // s3(t) ready / end-of-step
    }

    // acc/T = (sum_t s3) @ h2o^T / T + h2o_b.  Spike counts <= 98 are exact in bf16.
#pragma unroll
    for (int t = 0; t < 2; ++t)
#pragma unroll
        for (int j = 0; j < 8; ++j)
            S3[(j + 8 * g) * HDIM + (nl + 16 * t)] = (bf16)sum3[t][j];
    __syncthreads();

    if (tid < 160) {
        int mm = tid / 10, o = tid % 10;
        float acc = 0.f;
        for (int k = 0; k < HDIM; ++k)
            acc += (float)S3[mm * HDIM + k] * h2o_w[o * HDIM + k];
        out[(row0 + mm) * 10 + o] = acc * (1.0f / 98.0f) + h2o_b[o];
    }
}

extern "C" void kernel_launch(void* const* d_in, const int* in_sizes, int n_in,
                              void* d_out, int out_size, void* d_ws, size_t ws_size,
                              hipStream_t stream) {
    const float* x      = (const float*)d_in[0];
    const float* i2h1_w = (const float*)d_in[1];
    const float* i2h1_b = (const float*)d_in[2];
    const float* h2h1_w = (const float*)d_in[3];
    const float* h2h1_b = (const float*)d_in[4];
    const float* i2h2_w = (const float*)d_in[5];
    const float* i2h2_b = (const float*)d_in[6];
    const float* h2h2_w = (const float*)d_in[7];
    const float* h2h2_b = (const float*)d_in[8];
    const float* i2h3_w = (const float*)d_in[9];
    const float* i2h3_b = (const float*)d_in[10];
    const float* h2h3_w = (const float*)d_in[11];
    const float* h2h3_b = (const float*)d_in[12];
    const float* h2o_w  = (const float*)d_in[13];
    const float* h2o_b  = (const float*)d_in[14];
    const float* tadp1  = (const float*)d_in[15];
    const float* tadp2  = (const float*)d_in[16];
    const float* tadp3  = (const float*)d_in[17];
    const float* tm1    = (const float*)d_in[18];
    const float* tm2    = (const float*)d_in[19];
    const float* tm3    = (const float*)d_in[20];

    const int NW = HDIM * HDIM;                 // 262144 elems per weight matrix
    bf16* ws  = (bf16*)d_ws;                    // 5 * 512 KB = 2.5 MB of scratch
    bf16* Wh1 = ws + 0 * (size_t)NW;
    bf16* Wi2 = ws + 1 * (size_t)NW;
    bf16* Wh2 = ws + 2 * (size_t)NW;
    bf16* Wi3 = ws + 3 * (size_t)NW;
    bf16* Wh3 = ws + 4 * (size_t)NW;

    dim3 cb(256), cg((NW + 255) / 256);
    cvt_bf16_kernel<<<cg, cb, 0, stream>>>(h2h1_w, Wh1, NW);
    cvt_bf16_kernel<<<cg, cb, 0, stream>>>(i2h2_w, Wi2, NW);
    cvt_bf16_kernel<<<cg, cb, 0, stream>>>(h2h2_w, Wh2, NW);
    cvt_bf16_kernel<<<cg, cb, 0, stream>>>(i2h3_w, Wi3, NW);
    cvt_bf16_kernel<<<cg, cb, 0, stream>>>(h2h3_w, Wh3, NW);

    // 32 workgroups (16 batch rows each) x 512 threads (16 waves), persistent over all 98 steps
    srnn_alif_kernel<<<32, 512, 0, stream>>>(
        x, i2h1_w, i2h1_b, h2h1_b, i2h2_b, h2h2_b, i2h3_b, h2h3_b,
        h2o_w, h2o_b, tadp1, tadp2, tadp3, tm1, tm2, tm3,
        Wh1, Wi2, Wh2, Wi3, Wh3, (float*)d_out);
}
